// NSSTDecomposition_55886114456104
// MI455X (gfx1250) — matrix operand order repaired and checked by reading, not compiled, not run on hardware
//
#include <hip/hip_runtime.h>
#include <stdint.h>

typedef __attribute__((ext_vector_type(16))) __bf16 v16bf;
typedef __attribute__((ext_vector_type(8)))  float  v8f;
typedef __attribute__((ext_vector_type(8)))  unsigned int v8u;
typedef __attribute__((ext_vector_type(2)))  float  v2f;

constexpr int IMG_H  = 384;
constexpr int IMG_W  = 384;
constexpr int NIMG   = 24;          // 8 batches * 3 channels
constexpr int PIX    = IMG_H * IMG_W;
constexpr int NSC    = 3;
constexpr int KD     = 15;
constexpr int KTAPS  = 225;
constexpr int NSTEP  = 8;           // K = dy*16+dx, 16x16 grid = 256 = 8 * 32

// LDS tile: 8 output rows + 15 halo rows = 23 rows, 143 staged cols,
// pitch 144 bf16; copy B (shifted one element) only used by scale 2.
constexpr int TROWS  = 23;
constexpr int TPITCH = 144;          // bf16 elements per row
constexpr int TPW    = TPITCH / 2;   // u32 words per row = 72
constexpr int TCOLS  = 143;

// 5-tap Gaussian, sigma=1, normalized (matches reference numerically)
static __device__ const float G5[5] = {
    0.054488684f, 0.244201342f, 0.402619949f, 0.244201342f, 0.054488684f};

__device__ inline int refl(int i, int n) {
  i = (i < 0) ? -i : i;
  i = (i >= n) ? (2 * n - 2 - i) : i;
  return i;
}

__device__ inline unsigned short f2bf(float f) {
  unsigned int u = __builtin_bit_cast(unsigned int, f);
  unsigned int r = u + 0x7FFFu + ((u >> 16) & 1u);   // round-to-nearest-even
  return (unsigned short)(r >> 16);
}

// Write one A-matrix element: row M, K index, for given scale.
// 16-bit A layout used here (contiguous K, not the ISA table order): lanes 0-15 (M): VGPR0-7 = K0..15 (+32*t);
//                  lanes 16-31(M): VGPR0-7 = K16..31.
__device__ inline void write_a(unsigned short* Abuf, int scale, int M, int K,
                               unsigned short val) {
  int t = K >> 5, kk = K & 31;
  int lane, v;
  lane = (kk >> 4) * 16; v = (kk & 15) >> 1;
  lane += M;
  size_t idx = ((size_t)(scale * NSTEP + t) * 32 + (size_t)lane) * 16
               + (size_t)(v * 2 + (kk & 1));
  Abuf[idx] = val;
}

// ---------------------------------------------------------------------------
// Kernel 1: build oriented 15x15 filters on device, K = dy*16 + dx (row-padded
// to 16; pad positions stay zero from the memset). Shifted copies pack unused
// M rows: scale0 -> 4 dirs x shifts {(0,0),(0,1),(1,0),(1,1)};
//         scale1 -> 8 dirs x dx-shifts {0,1}; scale2 -> 16 dirs plain.
// A filter shifted +1 in dx computes pixel x+1; +1 in dy computes row y+1.
// ---------------------------------------------------------------------------
__global__ __launch_bounds__(256)
void prep_filters_kernel(unsigned short* __restrict__ Abuf) {
  int fb = blockIdx.x;      // 0..27 -> (scale, dir)
  int scale, dir, nd;
  if (fb < 4)       { scale = 0; dir = fb;      nd = 4;  }
  else if (fb < 12) { scale = 1; dir = fb - 4;  nd = 8;  }
  else              { scale = 2; dir = fb - 12; nd = 16; }

  int tid = threadIdx.x;
  int dy = tid / KD, dx = tid - dy * KD;

  float angle = 3.14159265358979323f * (float)dir / (float)nd;
  float sx = 2.0f * (float)(scale + 1);
  float sy = 0.5f * (float)(scale + 1);

  float val = 0.0f;
  if (tid < KTAPS) {
    float X = (float)(dx - 7);
    float Y = (float)(dy - 7);
    float ca = cosf(angle), sa = sinf(angle);
    float Xr = X * ca - Y * sa;
    float Yr = X * sa + Y * ca;
    val = expf(-0.5f * (Xr * Xr / (sx * sx) + Yr * Yr / (sy * sy))) * Xr / (sx * sx);
  }

  __shared__ float red[256];
  red[tid] = (tid < KTAPS) ? val : 0.0f;
  __syncthreads();
  for (int s = 128; s >= 1; s >>= 1) {
    if (tid < s) red[tid] += red[tid + s];
    __syncthreads();
  }
  float mean = red[0] / (float)KTAPS;
  __syncthreads();

  float vc = (tid < KTAPS) ? (val - mean) : 0.0f;
  red[tid] = vc * vc;
  __syncthreads();
  for (int s = 128; s >= 1; s >>= 1) {
    if (tid < s) red[tid] += red[tid + s];
    __syncthreads();
  }
  float nrm = sqrtf(red[0]);
  float inv = (nrm > 1e-6f) ? (1.0f / nrm) : 1.0f;

  if (tid < KTAPS) {
    unsigned short bv = f2bf(vc * inv);
    if (scale == 0) {
#pragma unroll
      for (int idx = 0; idx < 4; ++idx) {
        int shy = idx >> 1, shx = idx & 1;
        write_a(Abuf, 0, dir + 4 * idx, (dy + shy) * 16 + (dx + shx), bv);
      }
    } else if (scale == 1) {
#pragma unroll
      for (int shx = 0; shx < 2; ++shx)
        write_a(Abuf, 1, dir + 8 * shx, dy * 16 + (dx + shx), bv);
    } else {
      write_a(Abuf, 2, dir, dy * 16 + dx, bv);
    }
  }
}

// ---------------------------------------------------------------------------
// Kernel 2: one Laplacian-pyramid level: lp = gauss5x5(src) (reflect pad),
// band (bf16) = src - lp; lp goes to scratch (or, at the final level, straight
// into output channel 28 of each 29-channel group).
// ---------------------------------------------------------------------------
__global__ __launch_bounds__(256)
void lp_band_kernel(const float* __restrict__ src,
                    float* __restrict__ lp_out,
                    float* __restrict__ out_final,
                    unsigned short* __restrict__ band,
                    int final_stage) {
  int gid = blockIdx.x * blockDim.x + threadIdx.x;
  if (gid >= NIMG * PIX) return;
  int img = gid / PIX;
  int p   = gid - img * PIX;
  int y = p / IMG_W, x = p - y * IMG_W;
  const float* s = src + (size_t)img * PIX;

  float acc = 0.0f;
#pragma unroll
  for (int i = 0; i < 5; ++i) {
    int yy = refl(y + i - 2, IMG_H);
    float wy = G5[i];
#pragma unroll
    for (int j = 0; j < 5; ++j) {
      int xx = refl(x + j - 2, IMG_W);
      acc += wy * G5[j] * s[yy * IMG_W + xx];
    }
  }
  band[(size_t)img * PIX + p] = f2bf(s[p] - acc);
  if (final_stage) {
    int b = img / 3, c = img - 3 * b;
    out_final[(size_t)(b * 87 + c * 29 + 28) * PIX + p] = acc;
  } else {
    lp_out[(size_t)img * PIX + p] = acc;
  }
}

// ---------------------------------------------------------------------------
// Kernel 3: directional bank, implicit GEMM on V_WMMA_F32_16X16X32_BF16.
// Workgroup = 8 output rows x 128 pixels, one staged 23x143 bf16 tile.
// ---------------------------------------------------------------------------
__device__ inline v16bf load_a(const unsigned short* __restrict__ Abuf,
                               int scale, int t, int lane) {
  return *(const v16bf*)(Abuf + ((size_t)(scale * NSTEP + t) * 32 + lane) * 16);
}

__device__ inline v16bf load_b(const unsigned int* __restrict__ w, int idx) {
  v8u x;
#pragma unroll
  for (int v = 0; v < 8; ++v) x[v] = w[idx + v];
  return __builtin_bit_cast(v16bf, x);
}

__global__ __launch_bounds__(256)
void dirbank_wmma_kernel(const unsigned short* __restrict__ band,
                         const unsigned short* __restrict__ Abuf,
                         float* __restrict__ out) {
  __shared__ unsigned int ldsw[2 * TROWS * TPW];   // tileA | tileB (shifted)
  unsigned short* ldsh = (unsigned short*)ldsw;
  unsigned short* tA = ldsh;                       // element e -> src col e
  unsigned short* tB = ldsh + TROWS * TPITCH;      // element e -> src col e+1

  unsigned bid = blockIdx.x;
  int xblk = bid % 3;    bid /= 3;
  int yblk = bid % 48;   bid /= 48;
  int img  = bid % NIMG; bid /= NIMG;
  int scale = (int)bid;                            // 0..2
  int x0 = xblk * 128;
  int y0 = yblk * 8;

  const unsigned short* bsrc = band + ((size_t)scale * NIMG + img) * PIX;

  // cooperative reflect-padded stage: rows y0-7..y0+15, cols x0-7..x0+135
  for (int idx = threadIdx.x; idx < TROWS * TCOLS; idx += 256) {
    int r = idx / TCOLS, c = idx - r * TCOLS;
    int gy = refl(y0 - 7 + r, IMG_H);
    int gx = refl(x0 - 7 + c, IMG_W);
    unsigned short v = bsrc[gy * IMG_W + gx];
    tA[r * TPITCH + c] = v;
    if (scale == 2 && c > 0) tB[r * TPITCH + c - 1] = v;   // odd-pixel copy
  }
  __syncthreads();

  int wave  = threadIdx.x >> 5;
  int lane  = threadIdx.x & 31;
  int wrow0 = wave >> 2;             // 0..1
  int xw    = (wave & 3) * 32;       // 0,32,64,96
  int n     = lane & 15;             // B column; even pixel x0+xw+2n
  int lhigh = (lane >> 4) & 1;       // lanes 16-31 hold K 16..31 (next row)
  int colW  = (xw >> 1) + n;

  const unsigned int* wA = ldsw;
  const unsigned int* wB = ldsw + TROWS * TPW;

  int b  = img / 3, ch = img - 3 * b;
  float* obase = out + (size_t)(b * 87 + ch * 29) * PIX;
  int xbase = x0 + xw + 2 * n;

  if (scale == 2) {
    // 16 dirs: dual chain (even pixels from tileA, odd from shifted tileB)
    for (int rb = 0; rb < 4; ++rb) {
      int row = wrow0 + 2 * rb;
      int baseW = (row + lhigh) * TPW + colW;
      v8f c0 = {}, c1 = {};
#pragma unroll
      for (int t = 0; t < NSTEP; ++t) {
        v16bf a  = load_a(Abuf, 2, t, lane);
        v16bf b0 = load_b(wA, baseW + t * (2 * TPW));
        v16bf b1 = load_b(wB, baseW + t * (2 * TPW));
        c0 = __builtin_amdgcn_wmma_f32_16x16x32_bf16(false, a, false, b0,
                                                     (short)0, c0, false, false);
        c1 = __builtin_amdgcn_wmma_f32_16x16x32_bf16(false, a, false, b1,
                                                     (short)0, c1, false, false);
      }
      int y = y0 + row;
#pragma unroll
      for (int r = 0; r < 8; ++r) {
        int dir = (lane < 16) ? r : (r + 8);
        v2f val = {c0[r], c1[r]};
        *(v2f*)(obase + (size_t)(12 + dir) * PIX + (size_t)y * IMG_W + xbase) = val;
      }
    }
  } else if (scale == 1) {
    // 8 dirs x dx-shift{0,1}: one chain covers even+odd pixels
    for (int rb = 0; rb < 4; ++rb) {
      int row = wrow0 + 2 * rb;
      int baseW = (row + lhigh) * TPW + colW;
      v8f c = {};
#pragma unroll
      for (int t = 0; t < NSTEP; ++t) {
        v16bf a  = load_a(Abuf, 1, t, lane);
        v16bf b0 = load_b(wA, baseW + t * (2 * TPW));
        c = __builtin_amdgcn_wmma_f32_16x16x32_bf16(false, a, false, b0,
                                                    (short)0, c, false, false);
      }
      int y = y0 + row;
      int x = xbase + lhigh;                 // M>=8 rows are the +1-shifted dirs
#pragma unroll
      for (int r = 0; r < 8; ++r)
        obase[(size_t)(4 + r) * PIX + (size_t)y * IMG_W + x] = c[r];
    }
  } else {
    // 4 dirs x shifts {(0,0),(0,1),(1,0),(1,1)}: one chain covers a 2x2 block
    for (int rb = 0; rb < 2; ++rb) {
      int pair = wrow0 * 2 + rb;
      int row  = 2 * pair;
      int baseW = (row + lhigh) * TPW + colW;
      v8f c = {};
#pragma unroll
      for (int t = 0; t < NSTEP; ++t) {
        v16bf a  = load_a(Abuf, 0, t, lane);
        v16bf b0 = load_b(wA, baseW + t * (2 * TPW));
        c = __builtin_amdgcn_wmma_f32_16x16x32_bf16(false, a, false, b0,
                                                    (short)0, c, false, false);
      }
      int y = y0 + row + lhigh;              // M>=8 rows are the +1-dy dirs
#pragma unroll
      for (int r = 0; r < 8; ++r) {
        int dir = r & 3;
        int x   = xbase + (r >> 2);          // M rows 4-7 are the +1-dx dirs
        obase[(size_t)dir * PIX + (size_t)y * IMG_W + x] = c[r];
      }
    }
  }
}

// ---------------------------------------------------------------------------
// Launch
// ---------------------------------------------------------------------------
extern "C" void kernel_launch(void* const* d_in, const int* in_sizes, int n_in,
                              void* d_out, int out_size, void* d_ws, size_t ws_size,
                              hipStream_t stream) {
  (void)in_sizes; (void)n_in; (void)out_size; (void)ws_size;

  const float* x = (const float*)d_in[0];
  float* out = (float*)d_out;
  char*  ws  = (char*)d_ws;

  // workspace layout
  unsigned short* Abuf  = (unsigned short*)ws;                     //   24 KB
  unsigned short* bandv = (unsigned short*)(ws + 32768);           // 3*24*PIX bf16
  size_t band_bytes = (size_t)NSC * NIMG * PIX * sizeof(unsigned short);
  float* lpA = (float*)(ws + 32768 + band_bytes);                  // 24*PIX f32
  float* lpB = (float*)(ws + 32768 + band_bytes + (size_t)NIMG * PIX * sizeof(float));

  // zero filter buffer: pad columns/rows and unused K must be zero
  hipMemsetAsync(Abuf, 0, (size_t)NSC * NSTEP * 32 * 16 * sizeof(unsigned short), stream);

  hipLaunchKernelGGL(prep_filters_kernel, dim3(28), dim3(256), 0, stream, Abuf);

  int nb = (NIMG * PIX) / 256;
  hipLaunchKernelGGL(lp_band_kernel, dim3(nb), dim3(256), 0, stream,
                     x,   lpA, (float*)nullptr, bandv + 0 * (size_t)NIMG * PIX, 0);
  hipLaunchKernelGGL(lp_band_kernel, dim3(nb), dim3(256), 0, stream,
                     lpA, lpB, (float*)nullptr, bandv + 1 * (size_t)NIMG * PIX, 0);
  hipLaunchKernelGGL(lp_band_kernel, dim3(nb), dim3(256), 0, stream,
                     lpB, (float*)nullptr, out, bandv + 2 * (size_t)NIMG * PIX, 1);

  hipLaunchKernelGGL(dirbank_wmma_kernel, dim3(NSC * NIMG * 48 * 3), dim3(256),
                     0, stream, bandv, Abuf, out);
}
